// FlowFeature_53575422051036
// MI455X (gfx1250) — compile-verified, matrix operand order checked by reading, not run on hardware
//
#include <hip/hip_runtime.h>

// Problem constants (from reference): B=4, H=128, W=256, C=128, MD=4 -> 81 disp.
#define Bb 4
#define Hh 128
#define Ww 256
#define Cc 128
#define MD 4
#define ND 9        // 2*MD+1
#define NDISP 81    // ND*ND
#define FLOW_SCALE 0.05f

typedef __attribute__((ext_vector_type(16))) __bf16       v16bf;
typedef __attribute__((ext_vector_type(8)))  float        v8f;
typedef __attribute__((ext_vector_type(4)))  float        f4;
typedef __attribute__((ext_vector_type(4)))  unsigned int u4;
typedef __attribute__((ext_vector_type(4)))  __bf16       bf4;

// ---------------------------------------------------------------------------
// Pass 1: bilinear warp of cp_r by 0.05*flow, emitted as bf16 into workspace.
// One thread = one (pixel, 4-channel group). Gathers are near-coalesced since
// |flow*0.05| is sub-pixel, so the 4 taps share cachelines.
// ---------------------------------------------------------------------------
__global__ __launch_bounds__(256) void warp_bf16_kernel(
    const float* __restrict__ cp_r, const float* __restrict__ flow,
    __bf16* __restrict__ crw)
{
    const int CG  = Cc / 4;
    int idx = blockIdx.x * blockDim.x + threadIdx.x;
    int pix = idx / CG;
    int cg  = idx % CG;
    int b   = pix / (Hh * Ww);
    int rem = pix % (Hh * Ww);
    int y   = rem / Ww;
    int x   = rem % Ww;

    float fy = flow[2 * pix + 0] * FLOW_SCALE;
    float fx = flow[2 * pix + 1] * FLOW_SCALE;
    float qy = (float)y - fy;
    float qx = (float)x - fx;
    float fy0 = fminf(fmaxf(floorf(qy), 0.f), (float)(Hh - 2));
    float fx0 = fminf(fmaxf(floorf(qx), 0.f), (float)(Ww - 2));
    int   y0  = (int)fy0;
    int   x0  = (int)fx0;
    float ay  = fminf(fmaxf(qy - fy0, 0.f), 1.f);
    float ax  = fminf(fmaxf(qx - fx0, 0.f), 1.f);

    const float* base = cp_r + ((size_t)((b * Hh + y0) * Ww + x0)) * Cc + cg * 4;
    f4 tl = *(const f4*)(base);
    f4 tr = *(const f4*)(base + Cc);
    f4 bl = *(const f4*)(base + Ww * Cc);
    f4 br = *(const f4*)(base + Ww * Cc + Cc);
    f4 top = tl + ax * (tr - tl);
    f4 bot = bl + ax * (br - bl);
    f4 r   = top + ay * (bot - top);

    bf4 o;
    o[0] = (__bf16)r[0]; o[1] = (__bf16)r[1];
    o[2] = (__bf16)r[2]; o[3] = (__bf16)r[3];
    *(bf4*)(crw + (size_t)pix * Cc + cg * 4) = o;
}

// ---------------------------------------------------------------------------
// Pass 2: banded correlation via V_WMMA_F32_16X16X32_BF16.
// One wave32 per (b, y, 16-wide x tile). A = cl tile [16 x C] (bf16),
// B = warped-cr rows (bf16, from L2-resident workspace).
// 16-bit A/B lane layout (ISA 7.12.2): lane half h holds K = 8h+e (e<8) and
// K = 16+8h+(e-8) (e>=8) within each 32-wide K block -> two 8-element
// contiguous channel runs per lane per K block.
// Band extraction goes through a 16x32 LDS tile to avoid per-element
// exec-mask branch storms on the global stores.
// ---------------------------------------------------------------------------
__global__ __launch_bounds__(32) void corr_wmma_kernel(
    const float* __restrict__ cl, const __bf16* __restrict__ crw,
    float* __restrict__ out)
{
    __shared__ float tile[16][32];  // [M][j - (x0-8)] for one dy row-pair

    const int lane = threadIdx.x;   // 0..31, EXEC all ones (full wave)
    const int h    = lane >> 4;     // K half
    const int m    = lane & 15;     // A row (M) / B column (N)
    const int x0   = blockIdx.x * 16;
    const int y    = blockIdx.y;
    const int b    = blockIdx.z;

    // ---- A: cl[b, y, x0+m, :] -> 4 K-blocks of v16bf, f32 -> bf16 on the fly
    const float* ap = cl + ((size_t)((b * Hh + y) * Ww + (x0 + m))) * Cc;
    v16bf A[4];
#pragma unroll
    for (int kb = 0; kb < 4; ++kb) {
        const int c0 = 32 * kb + 8 * h;
        f4 r0 = *(const f4*)(ap + c0);
        f4 r1 = *(const f4*)(ap + c0 + 4);
        f4 r2 = *(const f4*)(ap + c0 + 16);
        f4 r3 = *(const f4*)(ap + c0 + 20);
#pragma unroll
        for (int t = 0; t < 4; ++t) {
            A[kb][t]      = (__bf16)r0[t];
            A[kb][4 + t]  = (__bf16)r1[t];
            A[kb][8 + t]  = (__bf16)r2[t];
            A[kb][12 + t] = (__bf16)r3[t];
        }
    }

    const float  scale  = 1.0f / (float)Cc;           // mean over channels
    const size_t outRow = ((size_t)((b * Hh + y) * Ww + x0)) * NDISP;

    for (int dyi = 0; dyi < ND; ++dyi) {
        int  yy  = y + dyi - MD;
        bool yok = (unsigned)yy < (unsigned)Hh;
        int  yc  = yok ? yy : 0;

#pragma unroll
        for (int jt = 0; jt < 2; ++jt) {
            int  j  = x0 - 8 + 16 * jt + m;           // column this lane feeds
            bool ok = yok && ((unsigned)j < (unsigned)Ww);
            int  jc = ((unsigned)j < (unsigned)Ww) ? j : 0;
            const __bf16* bp = crw + ((size_t)((b * Hh + yc) * Ww + jc)) * Cc;

            union BU { v16bf v; u4 q[2]; };
            BU Bm[4];
#pragma unroll
            for (int kb = 0; kb < 4; ++kb) {
                const int c0 = 32 * kb + 8 * h;
                u4 q0 = {0, 0, 0, 0}, q1 = {0, 0, 0, 0};
                if (ok) {                              // zero padding -> zero B
                    q0 = *(const u4*)(bp + c0);        // 8 bf16 = 16B
                    q1 = *(const u4*)(bp + c0 + 16);
                }
                Bm[kb].q[0] = q0;
                Bm[kb].q[1] = q1;
            }

            v8f acc = {};
#pragma unroll
            for (int kb = 0; kb < 4; ++kb) {
                acc = __builtin_amdgcn_wmma_f32_16x16x32_bf16(
                    /*neg_a=*/false, A[kb], /*neg_b=*/false, Bm[kb].v,
                    /*c_mod=*/(short)0, acc, /*reuse_a=*/false, /*reuse_b=*/false);
            }

            // Dump full D tile into LDS: lane holds D(M=i+8h, N=m).
#pragma unroll
            for (int i = 0; i < 8; ++i)
                tile[i + 8 * h][m + 16 * jt] = acc[i];
        }

        __syncthreads();   // single-wave group: just DS-counter ordering

        // Band read-out: lane (h,m) handles x = x0+m, dxi = 2t+h.
        // j = x + dxi - 4  ->  LDS column = m + dxi + 4 in [4, 27].
#pragma unroll
        for (int t = 0; t < 5; ++t) {
            int dxi = 2 * t + h;
            if (dxi < ND) {
                float v = tile[m][m + dxi + 4];
                out[outRow + (size_t)m * NDISP + dyi * ND + dxi] = v * scale;
            }
        }

        __syncthreads();   // protect tile before next dy overwrites it
    }
}

extern "C" void kernel_launch(void* const* d_in, const int* in_sizes, int n_in,
                              void* d_out, int out_size, void* d_ws, size_t ws_size,
                              hipStream_t stream) {
    (void)in_sizes; (void)n_in; (void)out_size; (void)ws_size;
    const float* cp_l = (const float*)d_in[0];
    const float* cp_r = (const float*)d_in[1];
    const float* flow = (const float*)d_in[2];
    float*   out = (float*)d_out;
    __bf16*  crw = (__bf16*)d_ws;   // 4*128*256*128 bf16 = 32 MB, L2-resident

    const int n1 = Bb * Hh * Ww * (Cc / 4);           // 4,194,304 threads
    warp_bf16_kernel<<<dim3(n1 / 256), dim3(256), 0, stream>>>(cp_r, flow, crw);
    corr_wmma_kernel<<<dim3(Ww / 16, Hh, Bb), dim3(32), 0, stream>>>(cp_l, crw, out);
}
